// MultiHeadAttentionPromotion_27324581937449
// MI455X (gfx1250) — compile-verified, matrix operand order checked by reading, not run on hardware
//
#include <hip/hip_runtime.h>
#include <hip/hip_bf16.h>

typedef __attribute__((ext_vector_type(16))) __bf16 v16bf;
typedef __attribute__((ext_vector_type(8)))  __bf16 v8bf;
typedef __attribute__((ext_vector_type(8)))  float  v8f;
typedef __attribute__((ext_vector_type(4)))  float  v4f;

constexpr int BS = 4, SL = 2048, IN_DIM = 1024, D_MODEL = 1024, H = 16, DEPTH = 64;
constexpr int MROWS = BS * SL; // 8192

// ---------------------------------------------------------------------------
// f32 -> bf16 conversion pre-pass (8 elements / thread, 16B stores).
// Streaming traffic: non-temporal on both sides.
// ---------------------------------------------------------------------------
__global__ __launch_bounds__(256) void cvt_kernel(
    const float* __restrict__ src, __bf16* __restrict__ dst)
{
    size_t base = ((size_t)blockIdx.x * 256 + threadIdx.x) * 8;
    const v4f* s4 = reinterpret_cast<const v4f*>(src + base);
    v4f f0 = __builtin_nontemporal_load(s4);
    v4f f1 = __builtin_nontemporal_load(s4 + 1);
    v8bf r;
    r[0] = (__bf16)f0[0]; r[1] = (__bf16)f0[1]; r[2] = (__bf16)f0[2]; r[3] = (__bf16)f0[3];
    r[4] = (__bf16)f1[0]; r[5] = (__bf16)f1[1]; r[6] = (__bf16)f1[2]; r[7] = (__bf16)f1[3];
    __builtin_nontemporal_store(r, reinterpret_cast<v8bf*>(dst + base));
}

// ---------------------------------------------------------------------------
// Projection GEMM (all-bf16 operands): dst = X @ W^T + bias, bf16 output in
// head-split layout (b,h,s,d); V stored transposed (b,h,d,s).
// Block = 256 threads (8 waves); each wave computes a 16x64 tile.
// ---------------------------------------------------------------------------
__global__ __launch_bounds__(256) void proj_kernel(
    const __bf16* __restrict__ X, const __bf16* __restrict__ W,
    const float* __restrict__ Bias, __bf16* __restrict__ dst, int vtrans)
{
    const int lane      = threadIdx.x & 31;
    const int wid       = threadIdx.x >> 5;
    const int colInTile = lane & 15;
    const int kHalf     = (lane >> 4) * 16;
    const int halfRow   = (lane >> 4) * 8;
    const int rowBase   = blockIdx.x * 16;
    const int colBase   = blockIdx.y * 512 + wid * 64;
    const int aRow      = rowBase + colInTile;

    v8f acc[4] = {};
    for (int kk = 0; kk < IN_DIM; kk += 32) {
        // Batch all operand loads for this K-step, then issue the 4 WMMAs.
        v16bf a = *(const v16bf*)(X + (size_t)aRow * IN_DIM + kk + kHalf);
        v16bf b[4];
#pragma unroll
        for (int nt = 0; nt < 4; ++nt) {
            int bRow = colBase + nt * 16 + colInTile; // W row == output column
            b[nt] = *(const v16bf*)(W + (size_t)bRow * IN_DIM + kk + kHalf);
        }
#pragma unroll
        for (int nt = 0; nt < 4; ++nt)
            acc[nt] = __builtin_amdgcn_wmma_f32_16x16x32_bf16(
                false, a, false, b[nt], (short)0, acc[nt], false, false);
    }

#pragma unroll
    for (int nt = 0; nt < 4; ++nt) {
        int col = colBase + nt * 16 + colInTile;
        float bv = Bias[col];
        int h = col >> 6, d = col & (DEPTH - 1);
#pragma unroll
        for (int r = 0; r < 8; ++r) {
            int row = rowBase + halfRow + r;
            int bb = row >> 11;            // row / SL
            int ss = row & (SL - 1);       // row % SL
            float v = acc[nt][r] + bv;
            size_t idx;
            if (vtrans) idx = (((size_t)(bb * H + h)) * DEPTH + d) * SL + ss;
            else        idx = (((size_t)(bb * H + h)) * SL + ss) * DEPTH + d;
            dst[idx] = (__bf16)v;
        }
    }
}

// ---------------------------------------------------------------------------
// Fused attention: per (b,h) head, per 64-row query block (4 waves x 16 rows).
// Two-pass online softmax with promotion matrix + causal mask; writes the
// full attention matrix (f32, non-temporal: pure streaming output) and
// accumulates O = P @ V via WMMA with a wave-private LDS transpose of P.
// ---------------------------------------------------------------------------
__global__ __launch_bounds__(128) void attn_kernel(
    const __bf16* __restrict__ qw, const __bf16* __restrict__ kw,
    const __bf16* __restrict__ vt, float* __restrict__ attn,
    __bf16* __restrict__ obuf)
{
    __shared__ __align__(64) __bf16 pbuf[4][16][64]; // 8 KB, wave-private strips

    const int lane      = threadIdx.x & 31;
    const int wid       = threadIdx.x >> 5;
    const int colInTile = lane & 15;
    const int kHalf     = (lane >> 4) * 16;
    const int halfRow   = (lane >> 4) * 8;
    const int bh        = blockIdx.y;          // b*H + h
    const int bx        = blockIdx.x;
    const int qBase     = bx * 64 + wid * 16;

    const __bf16* qrow = qw + (size_t)bh * SL * DEPTH;
    const __bf16* krow = kw + (size_t)bh * SL * DEPTH;
    const __bf16* vrow = vt + (size_t)bh * DEPTH * SL;
    float* attnRow = attn + (size_t)bh * SL * SL;

    // Q A-operand fragments for depth=64 (two K=32 chunks), reused all pass.
    v16bf aq[2];
#pragma unroll
    for (int t = 0; t < 2; ++t)
        aq[t] = *(const v16bf*)(qrow + (size_t)(qBase + colInTile) * DEPTH + t * 32 + kHalf);

    const float scale = 0.125f; // 1/sqrt(DEPTH)

    float mrun[8], srun[8];
#pragma unroll
    for (int r = 0; r < 8; ++r) { mrun[r] = -3.0e38f; srun[r] = 0.0f; }

    // ---------------- Pass 1: row max / sum (online softmax stats) ---------
    for (int kt = 0; kt <= bx; ++kt) {
        float xv[4][8];
#pragma unroll
        for (int nt = 0; nt < 4; ++nt) {
            int kn = kt * 64 + nt * 16 + colInTile;
            v8f s = {};
#pragma unroll
            for (int t = 0; t < 2; ++t) {
                v16bf b = *(const v16bf*)(krow + (size_t)kn * DEPTH + t * 32 + kHalf);
                s = __builtin_amdgcn_wmma_f32_16x16x32_bf16(
                    false, aq[t], false, b, (short)0, s, false, false);
            }
            int j = kn; // column index (lane-varying)
#pragma unroll
            for (int r = 0; r < 8; ++r) {
                int i = qBase + halfRow + r;
                xv[nt][r] = (j <= i)
                    ? s[r] * scale * __expf((float)(j + 1) / (float)(i + 1))
                    : -1.0e9f;
            }
        }
#pragma unroll
        for (int r = 0; r < 8; ++r) {
            float tm = fmaxf(fmaxf(xv[0][r], xv[1][r]), fmaxf(xv[2][r], xv[3][r]));
#pragma unroll
            for (int m = 1; m < 16; m <<= 1) tm = fmaxf(tm, __shfl_xor(tm, m, 32));
            float mnew = fmaxf(mrun[r], tm);
            float ps = __expf(xv[0][r] - mnew) + __expf(xv[1][r] - mnew) +
                       __expf(xv[2][r] - mnew) + __expf(xv[3][r] - mnew);
#pragma unroll
            for (int m = 1; m < 16; m <<= 1) ps += __shfl_xor(ps, m, 32);
            srun[r] = srun[r] * __expf(mrun[r] - mnew) + ps;
            mrun[r] = mnew;
        }
    }

    float inv[8];
#pragma unroll
    for (int r = 0; r < 8; ++r) inv[r] = 1.0f / srun[r];

    // ---------------- Pass 2: write attn, accumulate O = P @ V -------------
    v8f oacc[4] = {};
    for (int kt = 0; kt <= bx; ++kt) {
#pragma unroll
        for (int nt = 0; nt < 4; ++nt) {
            int kn = kt * 64 + nt * 16 + colInTile;
            v8f s = {};
#pragma unroll
            for (int t = 0; t < 2; ++t) {
                v16bf b = *(const v16bf*)(krow + (size_t)kn * DEPTH + t * 32 + kHalf);
                s = __builtin_amdgcn_wmma_f32_16x16x32_bf16(
                    false, aq[t], false, b, (short)0, s, false, false);
            }
            int j = kn;
#pragma unroll
            for (int r = 0; r < 8; ++r) {
                int i = qBase + halfRow + r;
                float v = (j <= i)
                    ? s[r] * scale * __expf((float)(j + 1) / (float)(i + 1))
                    : -1.0e9f;
                float p = __expf(v - mrun[r]) * inv[r]; // masked -> exact 0
                __builtin_nontemporal_store(p, attnRow + (size_t)i * SL + j);
                pbuf[wid][halfRow + r][nt * 16 + colInTile] = (__bf16)p;
            }
        }
        // O += P @ V ; A from wave-private LDS (transpose), B from Vt rows.
#pragma unroll
        for (int dt = 0; dt < 4; ++dt) {
#pragma unroll
            for (int t = 0; t < 2; ++t) {
                v16bf ap = *(const v16bf*)(&pbuf[wid][colInTile][t * 32 + kHalf]);
                v16bf bv = *(const v16bf*)(vrow + (size_t)(dt * 16 + colInTile) * SL
                                           + kt * 64 + t * 32 + kHalf);
                oacc[dt] = __builtin_amdgcn_wmma_f32_16x16x32_bf16(
                    false, ap, false, bv, (short)0, oacc[dt], false, false);
            }
        }
    }

    // Zero the never-touched (fully masked) columns of attn (d_out is poisoned).
    int jstart = (bx + 1) * 64;
    for (int r = 0; r < 16; ++r) {
        int i = qBase + r;
        for (int j = jstart + lane; j < SL; j += 32)
            __builtin_nontemporal_store(0.0f, attnRow + (size_t)i * SL + j);
    }

    // Write O to merged (b, s, h*DEPTH + d) bf16 buffer for the dense GEMM.
    int bb = bh >> 4;       // bh / H
    int hh = bh & (H - 1);  // bh % H
#pragma unroll
    for (int dt = 0; dt < 4; ++dt) {
        int d = dt * 16 + colInTile;
#pragma unroll
        for (int r = 0; r < 8; ++r) {
            int i = qBase + halfRow + r;
            obuf[(((size_t)bb * SL + i) * D_MODEL) + hh * DEPTH + d] = (__bf16)oacc[dt][r];
        }
    }
}

// ---------------------------------------------------------------------------
// Dense output projection (all-bf16 operands): out = O @ W^T + b (f32 output)
// ---------------------------------------------------------------------------
__global__ __launch_bounds__(256) void dense_kernel(
    const __bf16* __restrict__ O, const __bf16* __restrict__ W,
    const float* __restrict__ Bias, float* __restrict__ out)
{
    const int lane      = threadIdx.x & 31;
    const int wid       = threadIdx.x >> 5;
    const int colInTile = lane & 15;
    const int kHalf     = (lane >> 4) * 16;
    const int halfRow   = (lane >> 4) * 8;
    const int rowBase   = blockIdx.x * 16;
    const int colBase   = blockIdx.y * 512 + wid * 64;
    const int aRow      = rowBase + colInTile;

    v8f acc[4] = {};
    for (int kk = 0; kk < D_MODEL; kk += 32) {
        v16bf a = *(const v16bf*)(O + (size_t)aRow * D_MODEL + kk + kHalf);
        v16bf b[4];
#pragma unroll
        for (int nt = 0; nt < 4; ++nt) {
            int bRow = colBase + nt * 16 + colInTile;
            b[nt] = *(const v16bf*)(W + (size_t)bRow * D_MODEL + kk + kHalf);
        }
#pragma unroll
        for (int nt = 0; nt < 4; ++nt)
            acc[nt] = __builtin_amdgcn_wmma_f32_16x16x32_bf16(
                false, a, false, b[nt], (short)0, acc[nt], false, false);
    }

#pragma unroll
    for (int nt = 0; nt < 4; ++nt) {
        int col = colBase + nt * 16 + colInTile;
        float bv = Bias[col];
#pragma unroll
        for (int r = 0; r < 8; ++r) {
            int row = rowBase + halfRow + r;
            __builtin_nontemporal_store(acc[nt][r] + bv,
                                        out + (size_t)row * D_MODEL + col);
        }
    }
}

// ---------------------------------------------------------------------------
extern "C" void kernel_launch(void* const* d_in, const int* in_sizes, int n_in,
                              void* d_out, int out_size, void* d_ws, size_t ws_size,
                              hipStream_t stream)
{
    const float* q    = (const float*)d_in[0];
    const float* k    = (const float*)d_in[1];
    const float* v    = (const float*)d_in[2];
    // d_in[3] = mask (causal mask is computed analytically in-kernel)
    const float* wq_w = (const float*)d_in[4];
    const float* wq_b = (const float*)d_in[5];
    const float* wk_w = (const float*)d_in[6];
    const float* wk_b = (const float*)d_in[7];
    const float* wv_w = (const float*)d_in[8];
    const float* wv_b = (const float*)d_in[9];
    const float* dw   = (const float*)d_in[10];
    const float* db   = (const float*)d_in[11];

    char* ws = (char*)d_ws;
    const size_t actElems = (size_t)MROWS * IN_DIM;   // 8,388,608 (q/k/v, O)
    const size_t wElems   = (size_t)D_MODEL * IN_DIM; // 1,048,576 per weight
    size_t off = 0;
    __bf16* qb    = (__bf16*)(ws + off); off += actElems * 2;  // bf16 inputs
    __bf16* kb    = (__bf16*)(ws + off); off += actElems * 2;
    __bf16* vb    = (__bf16*)(ws + off); off += actElems * 2;
    __bf16* wqb   = (__bf16*)(ws + off); off += wElems * 2;    // bf16 weights
    __bf16* wkb   = (__bf16*)(ws + off); off += wElems * 2;
    __bf16* wvb   = (__bf16*)(ws + off); off += wElems * 2;
    __bf16* dwb   = (__bf16*)(ws + off); off += wElems * 2;
    __bf16* q_ws  = (__bf16*)(ws + off); off += actElems * 2;  // projected Q
    __bf16* k_ws  = (__bf16*)(ws + off); off += actElems * 2;  // projected K
    __bf16* vt_ws = (__bf16*)(ws + off); off += actElems * 2;  // projected V^T
    __bf16* o_ws  = (__bf16*)(ws + off); off += actElems * 2;  // attn output

    float* outp  = (float*)d_out;
    float* attnp = outp + (size_t)BS * SL * D_MODEL;

    // 1) bf16 conversion pre-pass (removes cvt chains from GEMM hot loops)
    cvt_kernel<<<dim3((unsigned)(actElems / 2048)), 256, 0, stream>>>(q, qb);
    cvt_kernel<<<dim3((unsigned)(actElems / 2048)), 256, 0, stream>>>(k, kb);
    cvt_kernel<<<dim3((unsigned)(actElems / 2048)), 256, 0, stream>>>(v, vb);
    cvt_kernel<<<dim3((unsigned)(wElems   / 2048)), 256, 0, stream>>>(wq_w, wqb);
    cvt_kernel<<<dim3((unsigned)(wElems   / 2048)), 256, 0, stream>>>(wk_w, wkb);
    cvt_kernel<<<dim3((unsigned)(wElems   / 2048)), 256, 0, stream>>>(wv_w, wvb);
    cvt_kernel<<<dim3((unsigned)(wElems   / 2048)), 256, 0, stream>>>(dw, dwb);

    // 2) QKV projections (bf16 WMMA)
    dim3 gProj(MROWS / 16, D_MODEL / 512);
    proj_kernel<<<gProj, 256, 0, stream>>>(qb, wqb, wq_b, q_ws, 0);
    proj_kernel<<<gProj, 256, 0, stream>>>(kb, wkb, wk_b, k_ws, 0);
    proj_kernel<<<gProj, 256, 0, stream>>>(vb, wvb, wv_b, vt_ws, 1);

    // 3) Fused promotion-softmax attention + O = P@V
    dim3 gAttn(SL / 64, BS * H);
    attn_kernel<<<gAttn, 128, 0, stream>>>(q_ws, k_ws, vt_ws, attnp, o_ws);

    // 4) Output projection
    dense_kernel<<<gProj, 256, 0, stream>>>(o_ws, dwb, db, outp);
}
